// DCAA_57604101374115
// MI455X (gfx1250) — matrix operand order repaired and checked by reading, not compiled, not run on hardware
//
#include <hip/hip_runtime.h>
#include <hip/hip_bf16.h>
#include <math.h>

typedef __attribute__((ext_vector_type(16))) _Float16 v16h;
typedef __attribute__((ext_vector_type(8)))  float    v8f;
typedef __attribute__((ext_vector_type(2)))  _Float16 h2f;

#define BATCH   16
#define CIN     64
#define HH      112
#define WW      112
#define HWSZ    (HH*WW)        // 12544
#define INIT_CH 64
#define NEW_CH  64
#define OUTC    128
#define NEXP    4
#define SEHID   32
#define EPSV    1e-5f

#define TILE_N  128            // pixels per block
#define BST     72             // padded LDS stride (halfs) for X tile rows (144B, 16B-aligned)

// ---------------------------------------------------------------------------
// Channel-mean pooling: dst[b*C+c] = mean_hw src[b*bstride + c*HWSZ + hw]
// ---------------------------------------------------------------------------
__global__ void pool_mean_kernel(const float* __restrict__ src,
                                 float* __restrict__ dst,
                                 int bstride, int C)
{
    int bc = blockIdx.x;                // b*C + c
    int b  = bc / C, c = bc % C;
    const float* p = src + (size_t)b * bstride + (size_t)c * HWSZ;
    float s = 0.f;
    for (int i = threadIdx.x; i < HWSZ; i += 256) s += p[i];
    __shared__ float red[256];
    red[threadIdx.x] = s;
    __syncthreads();
    for (int off = 128; off > 0; off >>= 1) {
        if (threadIdx.x < off) red[threadIdx.x] += red[threadIdx.x + off];
        __syncthreads();
    }
    if (threadIdx.x == 0) dst[bc] = red[0] * (1.0f / (float)HWSZ);
}

// ---------------------------------------------------------------------------
// Routing MLP 1 + expert mix -> per-sample 1x1 kernel K1[b,64,64] in f16
// ---------------------------------------------------------------------------
__global__ void mix1_kernel(const float* __restrict__ pooled,
                            const float* __restrict__ rw1,
                            const float* __restrict__ rb1,
                            const float* __restrict__ w1,
                            _Float16* __restrict__ k1h)
{
    __shared__ float r1[BATCH * NEXP];
    int t = threadIdx.x;
    if (t < BATCH * NEXP) {
        int b = t / NEXP, e = t % NEXP;
        float acc = rb1[e];
        for (int i = 0; i < CIN; ++i) acc += pooled[b*CIN + i] * rw1[e*CIN + i];
        r1[t] = 1.0f / (1.0f + expf(-acc));
    }
    __syncthreads();
    const int total = BATCH * INIT_CH * CIN;   // 65536
    for (int idx = t; idx < total; idx += 256) {
        int b  = idx >> 12;                    // /4096
        int oi = idx & 4095;                   // o*64+i
        float acc = 0.f;
#pragma unroll
        for (int e = 0; e < NEXP; ++e) acc += r1[b*NEXP + e] * w1[e*4096 + oi];
        k1h[idx] = (_Float16)acc;
    }
}

// ---------------------------------------------------------------------------
// Per-sample 1x1 conv as WMMA GEMM: X1[64,HW] = K1[64,64] x X[64,HW]
// fused BN1 + ReLU epilogue, writes channels [0,64) of d_out.
// Block = (sample b) x (128 pixels); 8 waves; wave w: Mtile = w>>1,
// Ntiles = {w&1, (w&1)+2, (w&1)+4, (w&1)+6}  -> 8 WMMAs per wave.
// LDS: K1 [o][k] (8KB), X tile transposed [px][ch] with pad (18KB) so that
// both A and B fragments are contiguous 16B chunks -> ds_load_b128.
// ---------------------------------------------------------------------------
__global__ void __launch_bounds__(256)
dynconv1x1_wmma_kernel(const float* __restrict__ x,
                       const _Float16* __restrict__ k1h,
                       const float* __restrict__ g, const float* __restrict__ bb,
                       const float* __restrict__ mm, const float* __restrict__ vv,
                       float* __restrict__ out)
{
    __shared__ __align__(16) _Float16 Ah[INIT_CH * CIN];   // K1: 64x64 f16
    __shared__ __align__(16) _Float16 Bh[TILE_N * BST];    // X tile: [px][ch] padded

    const int b   = blockIdx.y;
    const int hw0 = blockIdx.x * TILE_N;
    const int tid = threadIdx.x;

    // ---- K1[b] -> LDS (dword copies) ----
    {
        const uint32_t* src = (const uint32_t*)(k1h + (size_t)b * (INIT_CH * CIN));
        uint32_t* dst = (uint32_t*)Ah;
#pragma unroll
        for (int j = 0; j < 8; ++j) dst[tid + j*256] = src[tid + j*256];
    }
    // ---- X tile -> LDS f16 transposed [px][ch]; half2 (ch-pair) stores ----
    {
        const float4* xb4 = (const float4*)(x + (size_t)b * CIN * HWSZ + hw0);
        const int CH_F4 = HWSZ / 4;                  // float4 stride per channel
#pragma unroll
        for (int j = 0; j < 4; ++j) {
            int p   = tid + j*256;                   // 0..1023 (ch-pair x px4)
            int ch0 = (p >> 5) << 1;                 // 0,2,...,62
            int px4 = p & 31;                        // float4 index within 128 px
            float4 a4 = xb4[(size_t)ch0      * CH_F4 + px4];
            float4 b4 = xb4[(size_t)(ch0+1)  * CH_F4 + px4];
            // prefetch next pixel tile of this row (gfx1250 global_prefetch_b8)
            __builtin_prefetch((const void*)(xb4 + (size_t)ch0 * CH_F4 + px4 + TILE_N/4), 0, 3);
            float av[4] = {a4.x, a4.y, a4.z, a4.w};
            float bv[4] = {b4.x, b4.y, b4.z, b4.w};
            int px = px4 * 4;
#pragma unroll
            for (int i = 0; i < 4; ++i) {
                h2f hv; hv[0] = (_Float16)av[i]; hv[1] = (_Float16)bv[i];
                *(h2f*)(&Bh[(px + i) * BST + ch0]) = hv;
            }
        }
    }
    __syncthreads();

    const int wv   = tid >> 5, lane = tid & 31;
    const int mt   = wv >> 1,  ntp  = wv & 1;
    const int l15  = lane & 15;
    const int hi   = (lane >= 16);
    const int hi8  = hi ? 8 : 0;

    v8f c[4] = {v8f{}, v8f{}, v8f{}, v8f{}};
#pragma unroll
    for (int kk = 0; kk < 2; ++kk) {
        // A fragment: row (mt*16+l15), halfs: K = kk*32 + hi8 + {0..7, 16..23}
        union { v16h v; uint4 q[2]; } au;
        const _Float16* ap = &Ah[(mt*16 + l15) * CIN + kk*32 + hi8];
        au.q[0] = *(const uint4*)(ap);
        au.q[1] = *(const uint4*)(ap + 16);
#pragma unroll
        for (int t = 0; t < 4; ++t) {
            int nt = ntp + t*2;
            // B fragment: col (nt*16+l15), halfs: K = kk*32 + hi8 + {0..7, 16..23}
            union { v16h v; uint4 q[2]; } bu;
            const _Float16* bp = &Bh[(nt*16 + l15) * BST + kk*32 + hi8];
            bu.q[0] = *(const uint4*)(bp);
            bu.q[1] = *(const uint4*)(bp + 16);
            c[t] = __builtin_amdgcn_wmma_f32_16x16x32_f16(false, au.v, false, bu.v,
                                                          (short)0, c[t], false, false);
        }
    }

    // ---- Epilogue: BN1 + ReLU, store x1 into out channels [0,64) ----
#pragma unroll
    for (int r = 0; r < 8; ++r) {
        int o = mt*16 + r + hi8;
        float sc = g[o] * rsqrtf(vv[o] + EPSV);
        float sh = bb[o] - mm[o] * sc;
        float* orow = out + ((size_t)b * OUTC + o) * HWSZ + hw0 + l15;
#pragma unroll
        for (int t = 0; t < 4; ++t) {
            int nt = ntp + t*2;
            orow[nt*16] = fmaxf(c[t][r] * sc + sh, 0.0f);
        }
    }
}

// ---------------------------------------------------------------------------
// Routing MLP 2 + expert mix -> depthwise kernel k2[b,64,3,3] (BN2 scale folded)
// ---------------------------------------------------------------------------
__global__ void mix2_kernel(const float* __restrict__ pooled,
                            const float* __restrict__ rw2,
                            const float* __restrict__ rb2,
                            const float* __restrict__ w2,
                            const float* __restrict__ g2,
                            const float* __restrict__ v2,
                            float* __restrict__ k2)
{
    __shared__ float r2[BATCH * NEXP];
    int t = threadIdx.x;
    if (t < BATCH * NEXP) {
        int b = t / NEXP, e = t % NEXP;
        float acc = rb2[e];
        for (int i = 0; i < INIT_CH; ++i) acc += pooled[b*INIT_CH + i] * rw2[e*INIT_CH + i];
        r2[t] = 1.0f / (1.0f + expf(-acc));
    }
    __syncthreads();
    const int total = BATCH * NEW_CH * 9;       // 9216
    for (int idx = t; idx < total; idx += 256) {
        int b   = idx / (NEW_CH * 9);
        int rem = idx % (NEW_CH * 9);
        int o   = rem / 9;
        float acc = 0.f;
#pragma unroll
        for (int e = 0; e < NEXP; ++e) acc += r2[b*NEXP + e] * w2[e*(NEW_CH*9) + rem];
        float sc = g2[o] * rsqrtf(v2[o] + EPSV);
        k2[idx] = acc * sc;
    }
}

// ---------------------------------------------------------------------------
// Depthwise 3x3 (pad 1) + BN2 shift + ReLU; reads x1 from out[:,0:64],
// writes x2 into out[:,64:128].
// ---------------------------------------------------------------------------
__global__ void dwconv3x3_kernel(const float* __restrict__ k2,
                                 const float* __restrict__ g2,
                                 const float* __restrict__ b2,
                                 const float* __restrict__ m2,
                                 const float* __restrict__ v2,
                                 float* __restrict__ out)
{
    int idx = blockIdx.x * 256 + threadIdx.x;   // over B*NEW*HW
    if (idx >= BATCH * NEW_CH * HWSZ) return;
    int hw = idx % HWSZ;
    int bo = idx / HWSZ;                        // b*64 + o
    int o  = bo % NEW_CH, b = bo / NEW_CH;
    int h  = hw / WW,     w = hw % WW;

    const float* src = out + ((size_t)b * OUTC + o) * HWSZ;   // x1 channel o
    const float* kk  = k2 + (size_t)bo * 9;
    float sc  = g2[o] * rsqrtf(v2[o] + EPSV);
    float acc = b2[o] - m2[o] * sc;             // BN2 shift (scale folded in k2)
#pragma unroll
    for (int dh = -1; dh <= 1; ++dh) {
        int hh = h + dh;
        if (hh < 0 || hh >= HH) continue;
#pragma unroll
        for (int dw = -1; dw <= 1; ++dw) {
            int w2i = w + dw;
            if (w2i < 0 || w2i >= WW) continue;
            acc += src[hh*WW + w2i] * kk[(dh+1)*3 + (dw+1)];
        }
    }
    out[((size_t)b * OUTC + (INIT_CH + o)) * HWSZ + hw] = fmaxf(acc, 0.0f);
}

// ---------------------------------------------------------------------------
// SE MLP: means(128) -> relu(32) -> sigmoid(128); one block.
// ---------------------------------------------------------------------------
__global__ void se_kernel(const float* __restrict__ m1,   // x1 means [B,64]
                          const float* __restrict__ m2,   // x2 means [B,64]
                          const float* __restrict__ w1, const float* __restrict__ b1,
                          const float* __restrict__ w2, const float* __restrict__ b2,
                          float* __restrict__ s)
{
    __shared__ float msh[BATCH * OUTC];    // 2048
    __shared__ float hid[BATCH * SEHID];   // 512
    int t = threadIdx.x;
    for (int idx = t; idx < BATCH * OUTC; idx += 256) {
        int b = idx / OUTC, ch = idx % OUTC;
        msh[idx] = (ch < INIT_CH) ? m1[b*INIT_CH + ch]
                                  : m2[b*INIT_CH + ch - INIT_CH];
    }
    __syncthreads();
    for (int idx = t; idx < BATCH * SEHID; idx += 256) {
        int b = idx / SEHID, j = idx % SEHID;
        float acc = b1[j];
        for (int ch = 0; ch < OUTC; ++ch) acc += msh[b*OUTC + ch] * w1[j*OUTC + ch];
        hid[idx] = fmaxf(acc, 0.0f);
    }
    __syncthreads();
    for (int idx = t; idx < BATCH * OUTC; idx += 256) {
        int b = idx / OUTC, ch = idx % OUTC;
        float acc = b2[ch];
#pragma unroll
        for (int j = 0; j < SEHID; ++j) acc += hid[b*SEHID + j] * w2[ch*SEHID + j];
        s[idx] = 1.0f / (1.0f + expf(-acc));
    }
}

// ---------------------------------------------------------------------------
// In-place gating: out[b,ch,:] *= s[b,ch]   (float4 vectorized)
// ---------------------------------------------------------------------------
__global__ void scale_kernel(float* __restrict__ out, const float* __restrict__ s)
{
    size_t idx = (size_t)blockIdx.x * 256 + threadIdx.x;  // float4 index
    const size_t total4 = (size_t)BATCH * OUTC * HWSZ / 4;
    if (idx >= total4) return;
    int bc = (int)((idx * 4) / HWSZ);                     // b*128 + ch
    float4* p = (float4*)out + idx;
    float4 v = *p;
    float sv = s[bc];
    v.x *= sv; v.y *= sv; v.z *= sv; v.w *= sv;
    *p = v;
}

// ---------------------------------------------------------------------------
extern "C" void kernel_launch(void* const* d_in, const int* in_sizes, int n_in,
                              void* d_out, int out_size, void* d_ws, size_t ws_size,
                              hipStream_t stream)
{
    (void)in_sizes; (void)n_in; (void)out_size; (void)ws_size;

    const float* x     = (const float*)d_in[0];
    const float* rw1   = (const float*)d_in[1];
    const float* rb1   = (const float*)d_in[2];
    const float* w1    = (const float*)d_in[3];
    const float* bn1_g = (const float*)d_in[4];
    const float* bn1_b = (const float*)d_in[5];
    const float* bn1_m = (const float*)d_in[6];
    const float* bn1_v = (const float*)d_in[7];
    const float* rw2   = (const float*)d_in[8];
    const float* rb2   = (const float*)d_in[9];
    const float* w2    = (const float*)d_in[10];
    const float* bn2_g = (const float*)d_in[11];
    const float* bn2_b = (const float*)d_in[12];
    const float* bn2_m = (const float*)d_in[13];
    const float* bn2_v = (const float*)d_in[14];
    const float* se_w1 = (const float*)d_in[15];
    const float* se_b1 = (const float*)d_in[16];
    const float* se_w2 = (const float*)d_in[17];
    const float* se_b2 = (const float*)d_in[18];

    float* out = (float*)d_out;

    // workspace layout (floats)
    float* wsf      = (float*)d_ws;
    float* pooled0  = wsf;                 // [16,64]
    float* pooled1  = wsf + 1024;          // [16,64]  (x1 means, reused by SE)
    float* pooled2  = wsf + 2048;          // [16,64]  (x2 means)
    float* k2       = wsf + 4096;          // [16,64,9]
    float* sgate    = wsf + 16384;         // [16,128]
    _Float16* k1h   = (_Float16*)(wsf + 20480);  // [16,64,64] f16

    // 1) pool(x)
    pool_mean_kernel<<<BATCH * CIN, 256, 0, stream>>>(x, pooled0, CIN * HWSZ, CIN);
    // 2) routing + mix -> K1 (f16)
    mix1_kernel<<<1, 256, 0, stream>>>(pooled0, rw1, rb1, w1, k1h);
    // 3) WMMA GEMM + BN1 + ReLU -> out[:, 0:64]
    dynconv1x1_wmma_kernel<<<dim3(HWSZ / TILE_N, BATCH, 1), 256, 0, stream>>>(
        x, k1h, bn1_g, bn1_b, bn1_m, bn1_v, out);
    // 4) pool(x1)
    pool_mean_kernel<<<BATCH * INIT_CH, 256, 0, stream>>>(out, pooled1, OUTC * HWSZ, INIT_CH);
    // 5) routing + mix -> k2 (BN2 scale folded)
    mix2_kernel<<<1, 256, 0, stream>>>(pooled1, rw2, rb2, w2, bn2_g, bn2_v, k2);
    // 6) depthwise 3x3 + BN2 + ReLU -> out[:, 64:128]
    dwconv3x3_kernel<<<(BATCH * NEW_CH * HWSZ) / 256, 256, 0, stream>>>(
        k2, bn2_g, bn2_b, bn2_m, bn2_v, out);
    // 7) pool(x2)
    pool_mean_kernel<<<BATCH * NEW_CH, 256, 0, stream>>>(
        out + (size_t)INIT_CH * HWSZ, pooled2, OUTC * HWSZ, NEW_CH);
    // 8) SE gate
    se_kernel<<<1, 256, 0, stream>>>(pooled1, pooled2, se_w1, se_b1, se_w2, se_b2, sgate);
    // 9) in-place gating
    scale_kernel<<<((BATCH * OUTC * HWSZ / 4) + 255) / 256, 256, 0, stream>>>(out, sgate);
}
